// SequenceToSequenceEncoder_40596030881991
// MI455X (gfx1250) — compile-verified, matrix operand order checked by reading, not run on hardware
//
#include <hip/hip_runtime.h>
#include <hip/hip_bf16.h>
#include <math.h>

typedef __attribute__((ext_vector_type(16))) _Float16 v16h;
typedef __attribute__((ext_vector_type(8)))  _Float16 v8h;
typedef __attribute__((ext_vector_type(8)))  float    v8f;

#define GB 64      // batch
#define GT 512     // time
#define GDIN 512
#define GD 512
#define GN3 1536   // 3*D

// packed weight matrix: 96 n-tiles * 16 k-chunks * 32 lanes * 16 halfs
#define PACK_ELEMS (96u * 16u * 32u * 16u)         // 786432 halfs per matrix
#define XF16_BYTES ((size_t)GB * GT * GDIN * 2)    // 33554432
#define PACK_BYTES ((size_t)PACK_ELEMS * 2)        // 1572864 per matrix
#define HBUF_HALFS ((size_t)2 * 2 * GB * GD)       // dir x pingpong x 64 x 512

static __device__ inline v8f wmma16(v16h a, v16h b, v8f c) {
    return __builtin_amdgcn_wmma_f32_16x16x32_f16(
        /*neg_a=*/false, a, /*neg_b=*/false, b,
        /*c_mod=*/(short)0, c, /*reuse_a=*/false, /*reuse_b=*/false);
}

// A fragment: lane = (m = lane&15, kh = lane>>4); elems 0..7 = K[k0+8kh .. +7],
// elems 8..15 = K[k0+16+8kh .. +7]. Two aligned b128 loads.
static __device__ inline v16h load_afrag(const _Float16* __restrict__ row, int k0, int kh) {
    v8h lo = *reinterpret_cast<const v8h*>(row + k0 + 8 * kh);
    v8h hi = *reinterpret_cast<const v8h*>(row + k0 + 16 + 8 * kh);
    return __builtin_shufflevector(lo, hi, 0,1,2,3,4,5,6,7,8,9,10,11,12,13,14,15);
}

// B fragment from packed weights: fully contiguous 32B per lane.
static __device__ inline v16h load_bfrag(const _Float16* __restrict__ pack,
                                         int nTile, int kc, int lane) {
    const _Float16* p = pack + ((((size_t)nTile * 16 + kc) * 32 + lane) << 4);
    v8h lo = *reinterpret_cast<const v8h*>(p);
    v8h hi = *reinterpret_cast<const v8h*>(p + 8);
    return __builtin_shufflevector(lo, hi, 0,1,2,3,4,5,6,7,8,9,10,11,12,13,14,15);
}

__global__ void convert_x_f16(const float* __restrict__ x, _Float16* __restrict__ xf) {
    size_t i = ((size_t)blockIdx.x * blockDim.x + threadIdx.x) * 4;
    if (i + 3 < (size_t)GB * GT * GDIN) {
        float4 v = *reinterpret_cast<const float4*>(x + i);
        xf[i + 0] = (_Float16)v.x;
        xf[i + 1] = (_Float16)v.y;
        xf[i + 2] = (_Float16)v.z;
        xf[i + 3] = (_Float16)v.w;
    }
}

// Pack W/U [512 x 1536] fp32 row-major into WMMA B-fragment order (f16).
// matrix order in dst: 0=fwd_W, 1=fwd_U, 2=bwd_W, 3=bwd_U
__global__ void pack_weights(const float* __restrict__ Wf, const float* __restrict__ Uf,
                             const float* __restrict__ Wb, const float* __restrict__ Ub,
                             _Float16* __restrict__ dst) {
    int m = blockIdx.y;
    const float* src = (m == 0) ? Wf : (m == 1) ? Uf : (m == 2) ? Wb : Ub;
    _Float16* out = dst + (size_t)m * PACK_ELEMS;
    unsigned e = blockIdx.x * blockDim.x + threadIdx.x;
    if (e >= PACK_ELEMS) return;
    int i     = e & 15;
    int lane  = (e >> 4) & 31;
    int kc    = (e >> 9) & 15;
    int nTile = e >> 13;
    int kh = lane >> 4, n = lane & 15;
    int kl = (i < 8) ? (8 * kh + i) : (8 * kh + i + 8);
    int k  = kc * 32 + kl;
    out[e] = (_Float16)src[(size_t)k * GN3 + nTile * 16 + n];
}

__global__ void zero_hbuf(_Float16* __restrict__ h) {
    size_t i = (size_t)blockIdx.x * blockDim.x + threadIdx.x;
    if (i < HBUF_HALFS) h[i] = (_Float16)0.0f;
}

// One recurrence step, both directions. grid = (4 ntile-groups, 4 m-tiles, 2 dirs),
// block = 256 (8 waves); wave w handles n-tile = blockIdx.x*8 + w (16 output dims).
__global__ __launch_bounds__(256)
void gru_step(const _Float16* __restrict__ xf16,
              const _Float16* __restrict__ packs,
              const float* __restrict__ bf, const float* __restrict__ bb,
              _Float16* __restrict__ hbuf,
              float* __restrict__ out, int s) {
    const int wave = threadIdx.x >> 5;
    const int lane = threadIdx.x & 31;
    const int kh   = lane >> 4;
    const int ln15 = lane & 15;   // row m for A-fragment; col n for B/C/D

    const int nt  = blockIdx.x * 8 + wave;   // 0..31 over D
    const int b0  = blockIdx.y * 16;
    const int dir = blockIdx.z;
    const int t   = dir ? (GT - 1 - s) : s;  // bwd consumes/produces reversed time

    const _Float16* packW = packs + (size_t)(dir ? 2 : 0) * PACK_ELEMS;
    const _Float16* packU = packs + (size_t)(dir ? 3 : 1) * PACK_ELEMS;
    const float*    bias  = dir ? bb : bf;

    const _Float16* hR = hbuf + ((size_t)dir * 2 + (s & 1)) * (GB * GD);
    _Float16*       hW = hbuf + ((size_t)dir * 2 + ((s + 1) & 1)) * (GB * GD);

    const _Float16* xrow = xf16 + ((size_t)(b0 + ln15) * GT + t) * GDIN;
    const _Float16* hrow = hR + (size_t)(b0 + ln15) * GD;

    v8f cz = {}, cr = {}, chx = {}, chu = {};

    // x_t @ [Wz | Wr | Wh]
    #pragma unroll 4
    for (int kc = 0; kc < 16; ++kc) {
        v16h a  = load_afrag(xrow, kc * 32, kh);
        v16h wz = load_bfrag(packW, nt,      kc, lane);
        v16h wr = load_bfrag(packW, nt + 32, kc, lane);
        v16h wh = load_bfrag(packW, nt + 64, kc, lane);
        cz  = wmma16(a, wz, cz);
        cr  = wmma16(a, wr, cr);
        chx = wmma16(a, wh, chx);
    }
    // h_{t-1} @ [Uz | Ur | Uh]  (Uh part kept separate for r-gating)
    #pragma unroll 4
    for (int kc = 0; kc < 16; ++kc) {
        v16h a  = load_afrag(hrow, kc * 32, kh);
        v16h uz = load_bfrag(packU, nt,      kc, lane);
        v16h ur = load_bfrag(packU, nt + 32, kc, lane);
        v16h uh = load_bfrag(packU, nt + 64, kc, lane);
        cz  = wmma16(a, uz, cz);
        cr  = wmma16(a, ur, cr);
        chu = wmma16(a, uh, chu);
    }

    const int d = nt * 16 + ln15;
    const float bz = bias[d], brr = bias[GD + d], bh = bias[2 * GD + d];
    const int outOff = dir ? GD : 0;

    #pragma unroll
    for (int v = 0; v < 8; ++v) {
        const int bgl = b0 + v + 8 * kh;
        const float hp = (float)hR[(size_t)bgl * GD + d];
        float z = fminf(fmaxf(0.2f * (cz[v] + bz) + 0.5f, 0.0f), 1.0f);
        float r = fminf(fmaxf(0.2f * (cr[v] + brr) + 0.5f, 0.0f), 1.0f);
        float hh = tanhf(chx[v] + bh + r * chu[v]);
        float hn = z * hp + (1.0f - z) * hh;
        out[((size_t)bgl * GT + t) * (2 * GD) + outOff + d] = hn;
        hW[(size_t)bgl * GD + d] = (_Float16)hn;
    }
}

extern "C" void kernel_launch(void* const* d_in, const int* in_sizes, int n_in,
                              void* d_out, int out_size, void* d_ws, size_t ws_size,
                              hipStream_t stream) {
    const float* x  = (const float*)d_in[0];
    const float* fW = (const float*)d_in[1];
    const float* fU = (const float*)d_in[2];
    const float* fb = (const float*)d_in[3];
    const float* bW = (const float*)d_in[4];
    const float* bU = (const float*)d_in[5];
    const float* bb = (const float*)d_in[6];
    float* out = (float*)d_out;

    char* ws = (char*)d_ws;
    _Float16* xf16  = (_Float16*)ws;
    _Float16* packs = (_Float16*)(ws + XF16_BYTES);
    _Float16* hbuf  = (_Float16*)(ws + XF16_BYTES + 4 * PACK_BYTES);

    // prep (re-run every call; deterministic, re-initializes h state)
    convert_x_f16<<<(GB * GT * GDIN / 4 + 255) / 256, 256, 0, stream>>>(x, xf16);
    pack_weights<<<dim3((PACK_ELEMS + 255) / 256, 4), 256, 0, stream>>>(fW, fU, bW, bU, packs);
    zero_hbuf<<<(HBUF_HALFS + 255) / 256, 256, 0, stream>>>(hbuf);

    // sequential recurrence: one kernel per timestep (graph-captured)
    for (int s = 0; s < GT; ++s) {
        gru_step<<<dim3(4, 4, 2), 256, 0, stream>>>(xf16, packs, fb, bb, hbuf, out, s);
    }
}
